// LSTM_79190607003746
// MI455X (gfx1250) — matrix operand order repaired and checked by reading, not compiled, not run on hardware
//
#include <hip/hip_runtime.h>

// ---------------------------------------------------------------------------
// 3-layer LSTM (H=64) on gfx1250 using v_wmma_f32_16x16x32_bf16.
// One workgroup = 16 batch rows (one WMMA M tile), 4 waves, each wave owns a
// 16-wide h-chunk and its 4 gate tiles (i,f,g,o). Weights live in VGPRs for
// the whole time loop; h recirculates through double-buffered LDS; the x_t
// tile is software-pipelined (prefetched one step ahead) to hide global
// latency on the 512-step sequential recurrence.
// ---------------------------------------------------------------------------

typedef __bf16 bf16_t;
typedef bf16_t v16bf __attribute__((ext_vector_type(16)));
typedef float  v8f   __attribute__((ext_vector_type(8)));

#define TT  512
#define BB  2048
#define HID 64

__device__ __forceinline__ unsigned short f2bf(float f) {
  union { float f; unsigned u; } v; v.f = f;
  unsigned r = (v.u + 0x7FFFu + ((v.u >> 16) & 1u)) >> 16;  // RNE
  return (unsigned short)r;
}
__device__ __forceinline__ float bf2f(unsigned short h) {
  union { unsigned u; float f; } v; v.u = ((unsigned)h) << 16;
  return v.f;
}

__device__ __forceinline__ float fast_sigmoid(float x) {
#if __has_builtin(__builtin_amdgcn_exp2f) && __has_builtin(__builtin_amdgcn_rcpf)
  return __builtin_amdgcn_rcpf(1.0f + __builtin_amdgcn_exp2f(-1.442695041f * x));
#else
  return 1.0f / (1.0f + __expf(-x));
#endif
}
__device__ __forceinline__ float fast_tanh(float x) {
#if __has_builtin(__builtin_amdgcn_exp2f) && __has_builtin(__builtin_amdgcn_rcpf)
  return 2.0f * __builtin_amdgcn_rcpf(1.0f + __builtin_amdgcn_exp2f(-2.885390082f * x)) - 1.0f;
#else
  return tanhf(x);
#endif
}

union AFrag {            // one 16x32 bf16 WMMA operand fragment (8 VGPRs)
  v16bf v;
  unsigned short u[16];
  uint4 q[2];
};

// KSTEPS: number of K=32 WMMA steps over the combined [x | h] operand.
// LAYER0: Din=16 padded to 32 (kstep 0 = x, ksteps 1..2 = h).
// else  : Din=64 (ksteps 0..1 = x, ksteps 2..3 = h).
template <int KSTEPS, bool LAYER0>
__global__ __launch_bounds__(128) void lstm_layer_kernel(
    const float* __restrict__ x0,              // layer0 input [B][T][16] f32 (or null)
    const unsigned short* __restrict__ xin,    // prev layer out [T][B][64] bf16 (or null)
    const float* __restrict__ Wih,             // [256][Din]
    const float* __restrict__ Whh,             // [256][64]
    const float* __restrict__ bih,             // [256]
    const float* __restrict__ bhh,             // [256]
    unsigned short* __restrict__ out,          // [T][B][64] bf16, or [B][64] if !write_all
    const int write_all) {
  constexpr int DIN    = LAYER0 ? 16 : 64;
  constexpr int DINPAD = LAYER0 ? 32 : 64;

  const int tid  = threadIdx.x;
  const int wv   = tid >> 5;        // wave id 0..3 -> h columns wv*16..wv*16+15
  const int lane = tid & 31;
  const int ln   = lane & 15;
  const bool hi  = lane >= 16;
  const int bt16 = blockIdx.x * 16; // first batch row of this tile

  __shared__ __align__(16) unsigned short hbuf[2][16 * HID];

  // -------- stage weight (B-matrix) fragments into registers, once --------
  // Bc[k][n] = (k < DINPAD) ? W_ih[n][k] (zero-padded) : W_hh[n][k-DINPAD]
  // B layout (32x16 bf16): lane col n = L%16; elems 0..7 = K kb..kb+7,
  // elems 8..15 = K kb+16..kb+23, kb = ks*32 + (L<16 ? 0 : 8).
  AFrag Bf[4][KSTEPS];
  float bias[4];
  const int n_local = wv * 16 + ln;
#pragma unroll
  for (int g = 0; g < 4; ++g) {
    const int n = g * 64 + n_local;
    bias[g] = bih[n] + bhh[n];
#pragma unroll
    for (int ks = 0; ks < KSTEPS; ++ks) {
      const int kb = ks * 32 + (hi ? 8 : 0);
#pragma unroll
      for (int i = 0; i < 16; ++i) {
        const int k = kb + (i < 8 ? i : i + 8);
        float w;
        if (k < DINPAD) w = (k < DIN) ? Wih[n * DIN + k] : 0.0f;
        else            w = Whh[n * HID + (k - DINPAD)];
        Bf[g][ks].u[i] = f2bf(w);
      }
    }
  }

  // h_0 = 0
  for (int i = tid; i < 2 * 16 * HID; i += 128) ((unsigned short*)hbuf)[i] = 0;
  __syncthreads();

  v8f c;
#pragma unroll
  for (int e = 0; e < 8; ++e) c[e] = 0.0f;

  const int m = ln;  // A row = batch row within tile (same for lo/hi halves)

  // -------- x prefetch state (software pipeline, 1 step ahead) --------
  float4 xa, xb;        // LAYER0: raw fp32 row halves (converted at use time)
  AFrag  Axp[2];        // !LAYER0: bf16 fragments, directly usable
  if (LAYER0) {
    const float* xr = x0 + ((size_t)(bt16 + m) * TT + 0) * 16 + (hi ? 8 : 0);
    xa = *(const float4*)xr;
    xb = *(const float4*)(xr + 4);
  } else {
    const unsigned short* xr = xin + ((size_t)0 * BB + bt16 + m) * HID;
#pragma unroll
    for (int f = 0; f < 2; ++f) {
      const int cb = f * 32 + (hi ? 8 : 0);
      Axp[f].q[0] = *(const uint4*)(xr + cb);
      Axp[f].q[1] = *(const uint4*)(xr + cb + 16);
    }
  }

  int cur = 0;
  for (int t = 0; t < TT; ++t) {
    // ---- issue next-step x loads first (latency hidden behind this step) --
    const int tn = (t + 1 < TT) ? t + 1 : TT - 1;  // clamp: no OOB on last step
    float4 xan, xbn;
    AFrag  Axn[2];
    if (LAYER0) {
      const float* xr = x0 + ((size_t)(bt16 + m) * TT + tn) * 16 + (hi ? 8 : 0);
      xan = *(const float4*)xr;
      xbn = *(const float4*)(xr + 4);
    } else {
      const unsigned short* xr = xin + ((size_t)tn * BB + bt16 + m) * HID;
#pragma unroll
      for (int f = 0; f < 2; ++f) {
        const int cb = f * 32 + (hi ? 8 : 0);
        Axn[f].q[0] = *(const uint4*)(xr + cb);
        Axn[f].q[1] = *(const uint4*)(xr + cb + 16);
      }
    }

    // ---------------- build A fragments ([x_t | h_{t-1}], 16xK bf16) -------
    AFrag Af[KSTEPS];
    if (LAYER0) {
      // x frag (kstep 0): K 0..15 = x cols (convert prefetched fp32), pad 0
      Af[0].u[0] = f2bf(xa.x); Af[0].u[1] = f2bf(xa.y);
      Af[0].u[2] = f2bf(xa.z); Af[0].u[3] = f2bf(xa.w);
      Af[0].u[4] = f2bf(xb.x); Af[0].u[5] = f2bf(xb.y);
      Af[0].u[6] = f2bf(xb.z); Af[0].u[7] = f2bf(xb.w);
#pragma unroll
      for (int i = 8; i < 16; ++i) Af[0].u[i] = 0;
    } else {
#pragma unroll
      for (int f = 0; f < 2; ++f) Af[f] = Axp[f];
    }
    // h frags from LDS (row-major [16][64] bf16)
#pragma unroll
    for (int f = 0; f < 2; ++f) {
      const int cb = f * 32 + (hi ? 8 : 0);
      const unsigned short* hr = &hbuf[cur][m * HID + cb];
      Af[KSTEPS - 2 + f].q[0] = *(const uint4*)hr;
      Af[KSTEPS - 2 + f].q[1] = *(const uint4*)(hr + 16);
    }

    // ---------------- gates = A x Bc + bias via WMMA ------------------------
    v8f acc[4];
#pragma unroll
    for (int g = 0; g < 4; ++g) {
#pragma unroll
      for (int e = 0; e < 8; ++e) acc[g][e] = bias[g];
#pragma unroll
      for (int ks = 0; ks < KSTEPS; ++ks) {
        acc[g] = __builtin_amdgcn_wmma_f32_16x16x32_bf16(
            false, Af[ks].v, false, Bf[g][ks].v, (short)0, acc[g], false, false);
      }
    }

    // ---------------- LSTM cell (fp32, register-local per wave) -------------
    // C/D layout: elem e, lane L -> row M = e + 8*(L>=16), col N = L%16.
    const int nxt = cur ^ 1;
    unsigned short* hp = &hbuf[nxt][(hi ? 8 : 0) * HID + wv * 16 + ln];
#pragma unroll
    for (int e = 0; e < 8; ++e) {
      float ig = fast_sigmoid(acc[0][e]);
      float fg = fast_sigmoid(acc[1][e]);
      float gg = fast_tanh(acc[2][e]);
      float og = fast_sigmoid(acc[3][e]);
      float cn = fg * c[e] + ig * gg;
      c[e] = cn;
      hp[e * HID] = f2bf(og * fast_tanh(cn));
    }
    __syncthreads();

    // ---------------- stream h_t out, coalesced (128 x 16B) ----------------
    if (write_all || t == TT - 1) {
      const size_t obase = ((size_t)(write_all ? t : 0) * BB + bt16) * HID;
      const uint4* src = (const uint4*)hbuf[nxt];
      uint4* dst = (uint4*)(out + obase);
      dst[tid] = src[tid];  // 16 rows * 64 cols * 2B = 2048B = 128 uint4
    }

    // rotate prefetch registers
    if (LAYER0) { xa = xan; xb = xbn; }
    else {
#pragma unroll
      for (int f = 0; f < 2; ++f) Axp[f] = Axn[f];
    }
    cur = nxt;
  }
}

// Head: out[b][o] = h_last[b] . W_out[o] + b_out[o]   (2048 x 7, trivial)
__global__ void lstm_head_kernel(const unsigned short* __restrict__ hlast,
                                 const float* __restrict__ Wout,
                                 const float* __restrict__ bout,
                                 float* __restrict__ dout) {
  int idx = blockIdx.x * blockDim.x + threadIdx.x;
  if (idx >= BB * 7) return;
  int b = idx / 7, o = idx % 7;
  const unsigned short* hr = hlast + (size_t)b * HID;
  const float* wr = Wout + o * HID;
  float s = bout[o];
#pragma unroll 8
  for (int k = 0; k < HID; ++k) s += bf2f(hr[k]) * wr[k];
  dout[idx] = s;
}

extern "C" void kernel_launch(void* const* d_in, const int* in_sizes, int n_in,
                              void* d_out, int out_size, void* d_ws, size_t ws_size,
                              hipStream_t stream) {
  (void)in_sizes; (void)n_in; (void)out_size; (void)ws_size;
  const float* x = (const float*)d_in[0];
  const float* Wih[3] = {(const float*)d_in[1], (const float*)d_in[5], (const float*)d_in[9]};
  const float* Whh[3] = {(const float*)d_in[2], (const float*)d_in[6], (const float*)d_in[10]};
  const float* bih[3] = {(const float*)d_in[3], (const float*)d_in[7], (const float*)d_in[11]};
  const float* bhh[3] = {(const float*)d_in[4], (const float*)d_in[8], (const float*)d_in[12]};
  const float* Wout = (const float*)d_in[13];
  const float* bout = (const float*)d_in[14];

  // Workspace: two [T][B][64] bf16 ping-pong activation buffers + last-h.
  unsigned short* buf0  = (unsigned short*)d_ws;
  unsigned short* buf1  = buf0 + (size_t)TT * BB * HID;
  unsigned short* hlast = buf1 + (size_t)TT * BB * HID;

  dim3 grid(BB / 16), blk(128);
  lstm_layer_kernel<3, true ><<<grid, blk, 0, stream>>>(x,       nullptr, Wih[0], Whh[0], bih[0], bhh[0], buf0,  1);
  lstm_layer_kernel<4, false><<<grid, blk, 0, stream>>>(nullptr, buf0,    Wih[1], Whh[1], bih[1], bhh[1], buf1,  1);
  lstm_layer_kernel<4, false><<<grid, blk, 0, stream>>>(nullptr, buf1,    Wih[2], Whh[2], bih[2], bhh[2], hlast, 0);
  lstm_head_kernel<<<(BB * 7 + 255) / 256, 256, 0, stream>>>(hlast, Wout, bout, (float*)d_out);
}
